// GNNModel_85770496901266
// MI455X (gfx1250) — matrix operand order repaired and checked by reading, not compiled, not run on hardware
//
#include <hip/hip_runtime.h>

// ============================================================================
// GNN (NNConv x3 + BN + ReLU, mean-pool, 2-layer head) for MI455X (gfx1250).
//
// Roofline: dominant cost is the edge-MLP GEMM  h1[E,K] @ W2[K,K]  with
// K = 64*in_ch (2048 / 4096 / 4096)  ->  ~3.8 TFLOP total.  W2 in bf16 is
// 32 MB -> L2-resident (192 MB L2), so the GEMM is matrix-pipe bound; we run
// it through v_wmma_f32_16x16x32_bf16 (fp32 accumulate).  Materializing the
// per-edge weight tensor W_e (0.8 GB fp32) costs only ~70us of HBM traffic
// at 23.3 TB/s, so no fusion with the einsum is needed.
//
// GEMM data movement uses CDNA5 GLOBAL_LOAD_ASYNC_TO_LDS_B128 (ASYNCcnt)
// with double-buffered LDS tiles: async copies for tile k+1 are issued
// before the WMMA work on tile k, hiding global latency without staging
// through VGPRs.
//
// d_in layout assumption (jax pytree: dict-insertion order for top-level
// setup_inputs, sorted keys inside params):
//   0:x  1:edge_index[2,E]  2:edge_attr  3:batch
//   4:fc1_W 5:fc1_b 6:fc2_W 7:fc2_b
//   8+8L+{0..7}: layer L -> W1, W2, b1, b2, beta, bias, gamma, root
// ============================================================================

#define N_NODES  20000
#define N_EDGES  50000
#define N_GRAPHS 128
#define NODE_IN  32
#define EDGE_IN  16
#define HID      64
#define BN_EPS   1e-5f

typedef __attribute__((ext_vector_type(16))) __bf16 v16bf;
typedef __attribute__((ext_vector_type(8)))  float  v8f;

union FragAB { v16bf v; uint4 q[2]; };

__device__ __forceinline__ unsigned short f32_to_bf16(float f) {
  union { float f; unsigned u; } v; v.f = f;
  unsigned r = v.u + 0x7FFFu + ((v.u >> 16) & 1u);   // round-to-nearest-even
  return (unsigned short)(r >> 16);
}

// CDNA5 async global->LDS copy (16B per lane), tracked by ASYNCcnt.
// dst: LDS byte offset (flat-address low 32 bits of a __shared__ pointer),
// src: 64-bit global address.
__device__ __forceinline__ void async_copy_b128(unsigned lds_off,
                                                const void* gaddr) {
  asm volatile("global_load_async_to_lds_b128 %0, %1, off"
               :: "v"(lds_off), "v"((unsigned long long)(uintptr_t)gaddr)
               : "memory");
}
__device__ __forceinline__ void wait_async0() {
  asm volatile("s_wait_asynccnt 0x0" ::: "memory");
}
__device__ __forceinline__ unsigned lds_off_of(const void* p) {
  // LDS flat address = {SHARED_BASE, addr[31:0]} -> low 32 bits are the
  // LDS byte offset (ISA 10.2 aperture rules).
  return (unsigned)(uintptr_t)p;
}

// ---------------------------------------------------------------------------
__global__ void zero_f32(float* __restrict__ p, int n) {
  int i = blockIdx.x * 256 + threadIdx.x;
  if (i < n) p[i] = 0.0f;
}

// ---------------------------------------------------------------------------
// W2 [K,K] fp32 row-major -> W2T [N=K, K] bf16 (transposed), LDS-tiled so both
// global read and write are coalesced.
__global__ __launch_bounds__(256)
void cvt_transpose_bf16(const float* __restrict__ W, unsigned short* __restrict__ WT,
                        int K) {
  __shared__ float tile[32][33];
  const int bx = blockIdx.x;          // n tile
  const int by = blockIdx.y;          // k tile
  const int tx = threadIdx.x & 31;
  const int ty = threadIdx.x >> 5;    // 0..7
  #pragma unroll
  for (int j = 0; j < 4; ++j) {
    int r = ty + j * 8;               // k within tile
    tile[r][tx] = W[(size_t)(by * 32 + r) * K + bx * 32 + tx];
  }
  __syncthreads();
  #pragma unroll
  for (int j = 0; j < 4; ++j) {
    int r = ty + j * 8;               // n within tile
    WT[(size_t)(bx * 32 + r) * K + by * 32 + tx] = f32_to_bf16(tile[tx][r]);
  }
}

// ---------------------------------------------------------------------------
// h1[e, o] = relu(edge_attr[e,:] @ W1[:, o] + b1[o])   (bf16 output = GEMM A)
__global__ __launch_bounds__(256)
void edge_mlp1(const float* __restrict__ ea, const float* __restrict__ W1,
               const float* __restrict__ b1, unsigned short* __restrict__ H, int K) {
  __shared__ float e[EDGE_IN];
  const int eidx = blockIdx.x;
  if (threadIdx.x < EDGE_IN) e[threadIdx.x] = ea[eidx * EDGE_IN + threadIdx.x];
  __syncthreads();
  for (int o = threadIdx.x; o < K; o += 256) {
    float acc = b1[o];
    #pragma unroll
    for (int i = 0; i < EDGE_IN; ++i) acc = fmaf(e[i], W1[i * K + o], acc);
    H[(size_t)eidx * K + o] = f32_to_bf16(fmaxf(acc, 0.0f));
  }
}

// ---------------------------------------------------------------------------
// C[E,N] = A[E,K](bf16) @ B[K,N](bf16, given as BT[N,K]) + bias[N]
// Block tile 128x128, BK=32. 8 waves (wave32): wave (wm,wn) owns 32x64,
// i.e. 2x4 tiles of v_wmma_f32_16x16x32_bf16.  Double-buffered LDS fed by
// global_load_async_to_lds_b128; A rows are clamped (not predicated) since
// rows >= E are never stored.
__global__ __launch_bounds__(256)
void gemm_bf16_wmma(const unsigned short* __restrict__ A,
                    const unsigned short* __restrict__ BT,
                    const float* __restrict__ bias,
                    float* __restrict__ C,
                    int E, int K, int N) {
  __shared__ unsigned short As[2][128 * 32];   // row-major [row][k]
  __shared__ unsigned short Bs[2][128 * 32];   // col-major [col][k]

  const int t     = threadIdx.x;
  const int lane  = t & 31;
  const int wid   = t >> 5;
  const int wm    = (wid & 3) * 32;         // wave row base within tile
  const int wn    = (wid >> 2) * 64;        // wave col base within tile
  const int mbase = blockIdx.y * 128;
  const int nbase = blockIdx.x * 128;

  // Per-thread copy assignment: 512 chunks of 16B per 128x32 tile, 2 each
  // for A and B.  chunk -> (row r, k-quarter ch); second chunk is r + 64.
  const int r0 = t >> 2;
  const int ch = (t & 3) * 8;
  const int arow0 = min(mbase + r0,      E - 1);   // branch-free tail clamp
  const int arow1 = min(mbase + r0 + 64, E - 1);
  const unsigned short* gA0 = A  + (size_t)arow0 * K + ch;
  const unsigned short* gA1 = A  + (size_t)arow1 * K + ch;
  const unsigned short* gB0 = BT + (size_t)(nbase + r0)      * K + ch;
  const unsigned short* gB1 = BT + (size_t)(nbase + r0 + 64) * K + ch;
  unsigned ldsA0[2], ldsA1[2], ldsB0[2], ldsB1[2];
  #pragma unroll
  for (int b = 0; b < 2; ++b) {
    ldsA0[b] = lds_off_of(&As[b][r0 * 32 + ch]);
    ldsA1[b] = lds_off_of(&As[b][(r0 + 64) * 32 + ch]);
    ldsB0[b] = lds_off_of(&Bs[b][r0 * 32 + ch]);
    ldsB1[b] = lds_off_of(&Bs[b][(r0 + 64) * 32 + ch]);
  }

  v8f acc[2][4];
  #pragma unroll
  for (int i = 0; i < 2; ++i)
    #pragma unroll
    for (int j = 0; j < 4; ++j) {
      v8f z = {0.f, 0.f, 0.f, 0.f, 0.f, 0.f, 0.f, 0.f};
      acc[i][j] = z;
    }

  // Prologue: async-fill buffer 0 with K-tile 0.
  async_copy_b128(ldsA0[0], gA0);
  async_copy_b128(ldsA1[0], gA1);
  async_copy_b128(ldsB0[0], gB0);
  async_copy_b128(ldsB1[0], gB1);
  wait_async0();
  __syncthreads();

  const int nk = K / 32;
  int buf = 0;
  for (int i = 0; i < nk; ++i) {
    const int kb = i * 32;
    const bool more = (i + 1 < nk);
    if (more) {  // issue next tile into the other buffer; no wait yet
      const int kn = kb + 32;
      async_copy_b128(ldsA0[buf ^ 1], gA0 + kn);
      async_copy_b128(ldsA1[buf ^ 1], gA1 + kn);
      async_copy_b128(ldsB0[buf ^ 1], gB0 + kn);
      async_copy_b128(ldsB1[buf ^ 1], gB1 + kn);
    }

    // Fragment loads per CDNA5 ISA 16-bit layouts:
    //  A 16x32: lanes 0-15 = rows, halves {0..7}=K0..7, {8..15}=K16..23;
    //           lanes 16-31 same rows, K8..15 / K24..31.
    //  B 32x16: lanes 0-15 = cols, halves K0..7 / K16..23; lanes 16-31 K8..15 / K24..31.
    {
      const int kh  = lane >> 4;
      const int l15 = lane & 15;
      FragAB a[2], b[4];
      #pragma unroll
      for (int mt = 0; mt < 2; ++mt) {
        int row = wm + mt * 16 + l15;
        a[mt].q[0] = *(const uint4*)(&As[buf][row * 32 + kh * 8]);
        a[mt].q[1] = *(const uint4*)(&As[buf][row * 32 + 16 + kh * 8]);
      }
      #pragma unroll
      for (int nt = 0; nt < 4; ++nt) {
        int col = wn + nt * 16 + l15;
        b[nt].q[0] = *(const uint4*)(&Bs[buf][col * 32 + kh * 8]);
        b[nt].q[1] = *(const uint4*)(&Bs[buf][col * 32 + 16 + kh * 8]);
      }
      #pragma unroll
      for (int mt = 0; mt < 2; ++mt)
        #pragma unroll
        for (int nt = 0; nt < 4; ++nt)
          acc[mt][nt] = __builtin_amdgcn_wmma_f32_16x16x32_bf16(
              false, a[mt].v, false, b[nt].v, (short)0, acc[mt][nt],
              false, false);
    }

    if (more) wait_async0();   // our copies for tile i+1 landed
    __syncthreads();           // everyone's copies landed; LDS reads done
    buf ^= 1;
  }

  // Epilogue: C layout -> VGPR r: lanes 0-15 hold M=r, lanes 16-31 M=r+8.
  const int l15  = lane & 15;
  const int rsel = (lane < 16) ? 0 : 8;
  #pragma unroll
  for (int mt = 0; mt < 2; ++mt)
    #pragma unroll
    for (int nt = 0; nt < 4; ++nt) {
      int   col = nbase + wn + nt * 16 + l15;
      float bc  = bias[col];
      #pragma unroll
      for (int r = 0; r < 8; ++r) {
        int row = mbase + wm + mt * 16 + rsel + r;
        if (row < E) C[(size_t)row * N + col] = acc[mt][nt][r] + bc;
      }
    }
}

// ---------------------------------------------------------------------------
// msg[e,o] = sum_i x[src[e],i] * We[e, i*HID + o];  scatter-add into agg/cnt.
__global__ __launch_bounds__(256)
void msg_scatter(const float* __restrict__ We, const float* __restrict__ xin,
                 const int* __restrict__ ei, float* __restrict__ agg,
                 float* __restrict__ cnt, int in_ch, int K) {
  __shared__ float xs[4][HID];
  const int el = threadIdx.x >> 6;     // edge slot 0..3
  const int o  = threadIdx.x & 63;
  const int e  = blockIdx.x * 4 + el;
  if (e < N_EDGES) {
    int src = ei[e];
    if (o < in_ch) xs[el][o] = xin[(size_t)src * in_ch + o];
  }
  __syncthreads();
  if (e >= N_EDGES) return;
  const int dst = ei[N_EDGES + e];
  const float* w = We + (size_t)e * K;
  float acc = 0.0f;
  for (int i = 0; i < in_ch; ++i) acc = fmaf(xs[el][i], w[i * HID + o], acc);
  atomicAdd(&agg[(size_t)dst * HID + o], acc);
  if (o == 0) atomicAdd(&cnt[dst], 1.0f);
}

// ---------------------------------------------------------------------------
// h = agg/deg + x@root + bias; accumulate per-channel BN sum/sumsq via LDS.
__global__ __launch_bounds__(256)
void node_update(const float* __restrict__ agg, const float* __restrict__ cnt,
                 const float* __restrict__ xin, const float* __restrict__ root,
                 const float* __restrict__ bias, float* __restrict__ hnode,
                 float* __restrict__ bnsum, float* __restrict__ bnsq, int in_ch) {
  __shared__ float s1[HID], s2[HID];
  if (threadIdx.x < HID) { s1[threadIdx.x] = 0.0f; s2[threadIdx.x] = 0.0f; }
  __syncthreads();
  const int idx = blockIdx.x * 256 + threadIdx.x;   // node*64 + o
  const int n = idx >> 6;
  const int o = idx & 63;
  if (n < N_NODES) {
    float deg  = fmaxf(cnt[n], 1.0f);
    float mean = agg[idx] / deg;
    float dot = 0.0f;
    for (int i = 0; i < in_ch; ++i)
      dot = fmaf(xin[(size_t)n * in_ch + i], root[i * HID + o], dot);
    float h = mean + dot + bias[o];
    hnode[idx] = h;
    atomicAdd(&s1[o], h);          // ds_add_f32
    atomicAdd(&s2[o], h * h);
  }
  __syncthreads();
  if (threadIdx.x < HID) {
    atomicAdd(&bnsum[threadIdx.x], s1[threadIdx.x]);
    atomicAdd(&bnsq[threadIdx.x],  s2[threadIdx.x]);
  }
}

// ---------------------------------------------------------------------------
__global__ __launch_bounds__(256)
void bn_apply(const float* __restrict__ hnode, const float* __restrict__ bnsum,
              const float* __restrict__ bnsq, const float* __restrict__ gamma,
              const float* __restrict__ beta, float* __restrict__ xout) {
  const int idx = blockIdx.x * 256 + threadIdx.x;
  if (idx >= N_NODES * HID) return;
  const int o = idx & 63;
  const float invN = 1.0f / (float)N_NODES;
  float mu  = bnsum[o] * invN;
  float var = bnsq[o] * invN - mu * mu;
  float y = gamma[o] * (hnode[idx] - mu) * rsqrtf(var + BN_EPS) + beta[o];
  xout[idx] = fmaxf(y, 0.0f);
}

// ---------------------------------------------------------------------------
__global__ __launch_bounds__(256)
void pool_scatter(const float* __restrict__ x, const int* __restrict__ batch,
                  float* __restrict__ ps, float* __restrict__ pc) {
  const int idx = blockIdx.x * 256 + threadIdx.x;
  if (idx >= N_NODES * HID) return;
  const int n = idx >> 6, o = idx & 63;
  const int g = batch[n];
  atomicAdd(&ps[g * HID + o], x[idx]);
  if (o == 0) atomicAdd(&pc[g], 1.0f);
}

// ---------------------------------------------------------------------------
__global__ __launch_bounds__(128)
void head(const float* __restrict__ ps, const float* __restrict__ pc,
          const float* __restrict__ fc1W, const float* __restrict__ fc1b,
          const float* __restrict__ fc2W, const float* __restrict__ fc2b,
          float* __restrict__ out) {
  const int g = threadIdx.x;
  if (g >= N_GRAPHS) return;
  const float inv = 1.0f / fmaxf(pc[g], 1.0f);
  float gv[HID], hv[HID];
  for (int i = 0; i < HID; ++i) gv[i] = ps[g * HID + i] * inv;
  for (int o = 0; o < HID; ++o) {
    float a = fc1b[o];
    for (int i = 0; i < HID; ++i) a = fmaf(gv[i], fc1W[i * HID + o], a);
    hv[o] = fmaxf(a, 0.0f);
  }
  float r = fc2b[0];
  for (int i = 0; i < HID; ++i) r = fmaf(hv[i], fc2W[i], r);
  out[g] = r;
}

// ===========================================================================
extern "C" void kernel_launch(void* const* d_in, const int* in_sizes, int n_in,
                              void* d_out, int out_size, void* d_ws, size_t ws_size,
                              hipStream_t stream) {
  (void)in_sizes; (void)n_in; (void)out_size; (void)ws_size;

  const float* x0    = (const float*)d_in[0];
  const int*   ei    = (const int*)d_in[1];     // [2, N_EDGES] (src row, dst row)
  const float* ea    = (const float*)d_in[2];
  const int*   batch = (const int*)d_in[3];
  const float* fc1W  = (const float*)d_in[4];
  const float* fc1b  = (const float*)d_in[5];
  const float* fc2W  = (const float*)d_in[6];
  const float* fc2b  = (const float*)d_in[7];

  // ---- workspace carve-out (256B aligned), ~1.28 GB ----
  char* ws = (char*)d_ws;
  size_t off = 0;
  auto carve = [&](size_t bytes) -> void* {
    void* p = ws + off;
    off = (off + bytes + 255) & ~(size_t)255;
    return p;
  };
  unsigned short* W2T   = (unsigned short*)carve((size_t)4096 * 4096 * 2);
  unsigned short* H1    = (unsigned short*)carve((size_t)N_EDGES * 4096 * 2);
  float*          We    = (float*)carve((size_t)N_EDGES * 4096 * 4);
  float*          agg   = (float*)carve((size_t)N_NODES * HID * 4);
  float*          cnt   = (float*)carve((size_t)N_NODES * 4);
  float*          hnode = (float*)carve((size_t)N_NODES * HID * 4);
  float*          xcur  = (float*)carve((size_t)N_NODES * HID * 4);
  float*          bnsum = (float*)carve(HID * 4);
  float*          bnsq  = (float*)carve(HID * 4);
  float*          ps    = (float*)carve(N_GRAPHS * HID * 4);
  float*          pc    = (float*)carve(N_GRAPHS * 4);

  const float* xin = x0;
  for (int L = 0; L < 3; ++L) {
    const int in_ch = (L == 0) ? NODE_IN : HID;
    const int K = HID * in_ch;                 // 2048 / 4096 / 4096
    const float* W1    = (const float*)d_in[8 + L * 8 + 0];
    const float* W2    = (const float*)d_in[8 + L * 8 + 1];
    const float* b1    = (const float*)d_in[8 + L * 8 + 2];
    const float* b2    = (const float*)d_in[8 + L * 8 + 3];
    const float* beta  = (const float*)d_in[8 + L * 8 + 4];
    const float* bias  = (const float*)d_in[8 + L * 8 + 5];
    const float* gamma = (const float*)d_in[8 + L * 8 + 6];
    const float* root  = (const float*)d_in[8 + L * 8 + 7];

    zero_f32<<<(N_NODES * HID + 255) / 256, 256, 0, stream>>>(agg, N_NODES * HID);
    zero_f32<<<(N_NODES + 255) / 256, 256, 0, stream>>>(cnt, N_NODES);
    zero_f32<<<1, 256, 0, stream>>>(bnsum, HID);
    zero_f32<<<1, 256, 0, stream>>>(bnsq, HID);

    cvt_transpose_bf16<<<dim3(K / 32, K / 32), 256, 0, stream>>>(W2, W2T, K);
    edge_mlp1<<<N_EDGES, 256, 0, stream>>>(ea, W1, b1, H1, K);

    dim3 ggrid(K / 128, (N_EDGES + 127) / 128);
    gemm_bf16_wmma<<<ggrid, 256, 0, stream>>>(H1, W2T, b2, We, N_EDGES, K, K);

    msg_scatter<<<(N_EDGES + 3) / 4, 256, 0, stream>>>(We, xin, ei, agg, cnt, in_ch, K);
    node_update<<<(N_NODES * HID) / 256, 256, 0, stream>>>(agg, cnt, xin, root, bias,
                                                           hnode, bnsum, bnsq, in_ch);
    bn_apply<<<(N_NODES * HID) / 256, 256, 0, stream>>>(hnode, bnsum, bnsq, gamma,
                                                        beta, xcur);
    xin = xcur;
  }

  zero_f32<<<(N_GRAPHS * HID + 255) / 256, 256, 0, stream>>>(ps, N_GRAPHS * HID);
  zero_f32<<<1, 256, 0, stream>>>(pc, N_GRAPHS);
  pool_scatter<<<(N_NODES * HID) / 256, 256, 0, stream>>>(xcur, batch, ps, pc);
  head<<<1, 128, 0, stream>>>(ps, pc, fc1W, fc1b, fc2W, fc2b, (float*)d_out);
}
